// MoE_PyTorch_2001454760319
// MI455X (gfx1250) — matrix operand order repaired and checked by reading, not compiled, not run on hardware
//
#include <hip/hip_runtime.h>
#include <hip/hip_bf16.h>

#define NUM_EXPERTS 8
#define HID   1024
#define INTER 2048
#define NTOK  16384
#define BT    32
#define BLK_PER_E (NTOK / BT)       /* 512 */
#define HROW  (INTER + 8)           /* padded LDS row, ushort elems */

typedef __attribute__((ext_vector_type(16))) __bf16 v16bf;
typedef __attribute__((ext_vector_type(8)))  float  v8f;

union Frag { uint4 q[2]; v16bf v; };

__device__ __forceinline__ unsigned short f2bf(float f) {
  unsigned u = __float_as_uint(f);
  u += 0x7FFFu + ((u >> 16) & 1u);           // round-to-nearest-even
  return (unsigned short)(u >> 16);
}

// ---------------------------------------------------------------- utils
__global__ void zero_counts_kernel(int* counts) {
  if (threadIdx.x < NUM_EXPERTS) counts[threadIdx.x] = 0;
}

__global__ void cvt_bf16_kernel(const float* __restrict__ src,
                                unsigned short* __restrict__ dst, int n4) {
  const float4* s4 = (const float4*)src;
  ushort4* d4 = (ushort4*)dst;
  for (int i = blockIdx.x * blockDim.x + threadIdx.x; i < n4;
       i += gridDim.x * blockDim.x) {
    float4 f = s4[i];
    ushort4 o;
    o.x = f2bf(f.x); o.y = f2bf(f.y); o.z = f2bf(f.z); o.w = f2bf(f.w);
    d4[i] = o;
  }
}

// ---------------------------------------------------------------- router
__global__ __launch_bounds__(256) void router_kernel(
    const float* __restrict__ x, const float* __restrict__ gw,
    int* counts, int* tok_list, float* wgt_list, float* logits_out) {
  __shared__ float gws[NUM_EXPERTS * HID];             // 32 KB
  for (int i = threadIdx.x; i < NUM_EXPERTS * HID; i += 256) gws[i] = gw[i];
  __syncthreads();

  const int wid  = threadIdx.x >> 5;
  const int lane = threadIdx.x & 31;
  const int t = blockIdx.x * 8 + wid;
  if (t >= NTOK) return;

  float acc[NUM_EXPERTS];
  #pragma unroll
  for (int e = 0; e < NUM_EXPERTS; ++e) acc[e] = 0.0f;

  const float* xr = x + (size_t)t * HID;
  for (int k = lane; k < HID; k += 32) {
    float xv = xr[k];
    #pragma unroll
    for (int e = 0; e < NUM_EXPERTS; ++e) acc[e] += xv * gws[e * HID + k];
  }
  #pragma unroll
  for (int e = 0; e < NUM_EXPERTS; ++e) {
    #pragma unroll
    for (int off = 16; off > 0; off >>= 1) acc[e] += __shfl_xor(acc[e], off, 32);
  }
  if (lane == 0) {
    #pragma unroll
    for (int e = 0; e < NUM_EXPERTS; ++e)
      logits_out[(size_t)t * NUM_EXPERTS + e] = acc[e];
    int i1 = 0; float l1 = acc[0];
    #pragma unroll
    for (int e = 1; e < NUM_EXPERTS; ++e) if (acc[e] > l1) { l1 = acc[e]; i1 = e; }
    int i2 = (i1 == 0) ? 1 : 0; float l2 = acc[i2];
    #pragma unroll
    for (int e = 0; e < NUM_EXPERTS; ++e)
      if (e != i1 && acc[e] > l2) { l2 = acc[e]; i2 = e; }
    float e2 = __expf(l2 - l1);
    float p1 = 1.0f / (1.0f + e2);
    float p2 = e2 / (1.0f + e2);
    int pos1 = atomicAdd(&counts[i1], 1);
    tok_list[i1 * NTOK + pos1] = t;                    // slot 0
    wgt_list[i1 * NTOK + pos1] = p1;
    int pos2 = atomicAdd(&counts[i2], 1);
    tok_list[i2 * NTOK + pos2] = t | (int)0x80000000;  // slot 1
    wgt_list[i2 * NTOK + pos2] = p2;
  }
}

// ---------------------------------------------------------------- fused expert
// Block: 256 threads = 8 waves; each wave computes a 32M x 64N tile
// (2 A-fragments x 4 B-fragments -> 8 WMMA accumulators); waves split N.
__global__ __launch_bounds__(256) void moe_expert_kernel(
    const unsigned short* __restrict__ xb,     // [T,H]   bf16
    const unsigned short* __restrict__ w1b,    // [E,I,H] bf16
    const unsigned short* __restrict__ w2b,    // [E,H,I] bf16
    const float* __restrict__ b1,              // [E,I]
    const float* __restrict__ b2,              // [E,H]
    const int*   __restrict__ counts,
    const int*   __restrict__ tok_list,        // [E,T] token | slot<<31
    const float* __restrict__ wgt_list,        // [E,T]
    float* __restrict__ y_slot) {              // [2,T,H]
  extern __shared__ char smem[];
  unsigned short (*hbuf)[HROW] = (unsigned short (*)[HROW])smem;
  int*   tokS  = (int*)(smem + (size_t)BT * HROW * 2);
  int*   slotS = tokS + BT;
  float* wS    = (float*)(slotS + BT);

  const int e    = blockIdx.x / BLK_PER_E;
  const int blk  = blockIdx.x % BLK_PER_E;
  const int nTok = counts[e];
  const int t0   = blk * BT;
  if (t0 >= nTok) return;

  const int tid = threadIdx.x;
  if (tid < BT) {
    int g = t0 + tid;
    if (g < nTok) {
      unsigned p = (unsigned)tok_list[e * NTOK + g];
      tokS[tid]  = (int)(p & 0x7FFFFFFFu);
      slotS[tid] = (int)(p >> 31);
      wS[tid]    = wgt_list[e * NTOK + g];
    } else {
      tokS[tid] = 0; slotS[tid] = -1; wS[tid] = 0.0f;   // padded row: skip store
    }
  }
  __syncthreads();

  const int lane = tid & 31;
  const int wid  = tid >> 5;
  const int half = lane >> 4;        // lane group 0-15 / 16-31
  const int l15  = lane & 15;
  const int wn   = wid * 64;         // 8 waves split N

  // A row base pointers (x, bf16): A layout has M = lane&15 for both halves
  const char* paBase0 = (const char*)(xb + (size_t)tokS[l15] * HID);
  const char* paBase1 = (const char*)(xb + (size_t)tokS[16 + l15] * HID);

  // ---------- Phase 1: h = gelu(x @ w1^T + b1), full INTER into LDS ----------
  for (int n0 = 0; n0 < INTER; n0 += 512) {
    const int nA = n0 + wn;
    v8f c[2][4] = {};
    const char* pb[4];
    #pragma unroll
    for (int i = 0; i < 4; ++i)
      pb[i] = (const char*)(w1b +
              ((size_t)e * INTER + (size_t)(nA + i * 16 + l15)) * HID + half * 8);
    for (int kk = 0; kk < HID; kk += 32) {
      Frag a0, a1;
      a0.q[0] = *(const uint4*)(paBase0 + kk * 2 + half * 16);
      a0.q[1] = *(const uint4*)(paBase0 + kk * 2 + 32 + half * 16);
      a1.q[0] = *(const uint4*)(paBase1 + kk * 2 + half * 16);
      a1.q[1] = *(const uint4*)(paBase1 + kk * 2 + 32 + half * 16);
      #pragma unroll
      for (int i = 0; i < 4; ++i) {
        Frag b;
        b.q[0] = *(const uint4*)(pb[i] + kk * 2);
        b.q[1] = *(const uint4*)(pb[i] + kk * 2 + 32);
        c[0][i] = __builtin_amdgcn_wmma_f32_16x16x32_bf16(
                    false, a0.v, false, b.v, (short)0, c[0][i], false, false);
        c[1][i] = __builtin_amdgcn_wmma_f32_16x16x32_bf16(
                    false, a1.v, false, b.v, (short)0, c[1][i], false, false);
      }
    }
    #pragma unroll
    for (int mi = 0; mi < 2; ++mi) {
      #pragma unroll
      for (int i = 0; i < 4; ++i) {
        const int n = nA + i * 16 + l15;
        const float bias = b1[e * INTER + n];
        #pragma unroll
        for (int r = 0; r < 8; ++r) {           // C layout: M = mi*16 + half*8 + r
          float v = c[mi][i][r] + bias;
          v = 0.5f * v * (1.0f + erff(v * 0.70710678118654752f));  // exact gelu
          hbuf[mi * 16 + half * 8 + r][n] = f2bf(v);
        }
      }
    }
  }
  __syncthreads();

  // ---------- Phase 2: y = (h @ w2^T + b2) * w, scatter to slot buffer ----------
  const char* phBase0 = (const char*)(&hbuf[l15][0]);
  const char* phBase1 = (const char*)(&hbuf[16 + l15][0]);
  for (int n0 = 0; n0 < HID; n0 += 512) {
    const int nA = n0 + wn;
    v8f c[2][4] = {};
    const char* pb[4];
    #pragma unroll
    for (int i = 0; i < 4; ++i)
      pb[i] = (const char*)(w2b +
              ((size_t)e * HID + (size_t)(nA + i * 16 + l15)) * INTER + half * 8);
    for (int kk = 0; kk < INTER; kk += 32) {
      Frag a0, a1;                               // A from LDS (ds_load_b128)
      a0.q[0] = *(const uint4*)(phBase0 + kk * 2 + half * 16);
      a0.q[1] = *(const uint4*)(phBase0 + kk * 2 + 32 + half * 16);
      a1.q[0] = *(const uint4*)(phBase1 + kk * 2 + half * 16);
      a1.q[1] = *(const uint4*)(phBase1 + kk * 2 + 32 + half * 16);
      #pragma unroll
      for (int i = 0; i < 4; ++i) {
        Frag b;
        b.q[0] = *(const uint4*)(pb[i] + kk * 2);
        b.q[1] = *(const uint4*)(pb[i] + kk * 2 + 32);
        c[0][i] = __builtin_amdgcn_wmma_f32_16x16x32_bf16(
                    false, a0.v, false, b.v, (short)0, c[0][i], false, false);
        c[1][i] = __builtin_amdgcn_wmma_f32_16x16x32_bf16(
                    false, a1.v, false, b.v, (short)0, c[1][i], false, false);
      }
    }
    #pragma unroll
    for (int mi = 0; mi < 2; ++mi) {
      #pragma unroll
      for (int i = 0; i < 4; ++i) {
        const int n = nA + i * 16 + l15;
        const float bias = b2[e * HID + n];
        #pragma unroll
        for (int r = 0; r < 8; ++r) {
          const int m = mi * 16 + half * 8 + r;
          const int slot = slotS[m];
          if (slot >= 0) {
            float v = (c[mi][i][r] + bias) * wS[m];
            y_slot[(size_t)slot * NTOK * HID + (size_t)tokS[m] * HID + n] = v;
          }
        }
      }
    }
  }
}

// ---------------------------------------------------------------- finalize
__global__ void finalize_kernel(const float* __restrict__ y_slot,
                                float* __restrict__ out) {
  const size_t n4 = (size_t)NTOK * HID / 4;
  const float4* y0 = (const float4*)y_slot;
  const float4* y1 = (const float4*)(y_slot + (size_t)NTOK * HID);
  float4* o = (float4*)out;
  for (size_t i = (size_t)blockIdx.x * blockDim.x + threadIdx.x; i < n4;
       i += (size_t)gridDim.x * blockDim.x) {
    float4 a = y0[i], b = y1[i];
    o[i] = make_float4(a.x + b.x, a.y + b.y, a.z + b.z, a.w + b.w);
  }
}

// ---------------------------------------------------------------- launch
extern "C" void kernel_launch(void* const* d_in, const int* in_sizes, int n_in,
                              void* d_out, int out_size, void* d_ws, size_t ws_size,
                              hipStream_t stream) {
  const float* x  = (const float*)d_in[0];   // [4,4096,1024]
  const float* gw = (const float*)d_in[1];   // [8,1024]
  const float* w1 = (const float*)d_in[2];   // [8,2048,1024]
  const float* b1 = (const float*)d_in[3];   // [8,2048]
  const float* w2 = (const float*)d_in[4];   // [8,1024,2048]
  const float* b2 = (const float*)d_in[5];   // [8,1024]

  float* out = (float*)d_out;                          // tuple: out then logits
  float* logits_out = out + (size_t)NTOK * HID;

  char* ws = (char*)d_ws;
  size_t off = 0;
  auto carve = [&](size_t bytes) -> char* {
    char* p = ws + off;
    off += (bytes + 255) & ~(size_t)255;
    return p;
  };
  unsigned short* xb  = (unsigned short*)carve((size_t)NTOK * HID * 2);
  unsigned short* w1b = (unsigned short*)carve((size_t)NUM_EXPERTS * INTER * HID * 2);
  unsigned short* w2b = (unsigned short*)carve((size_t)NUM_EXPERTS * HID * INTER * 2);
  float* y_slot   = (float*)carve((size_t)2 * NTOK * HID * 4);
  int*   tok_list = (int*)carve((size_t)NUM_EXPERTS * NTOK * 4);
  float* wgt_list = (float*)carve((size_t)NUM_EXPERTS * NTOK * 4);
  int*   counts   = (int*)carve(256);

  zero_counts_kernel<<<1, 32, 0, stream>>>(counts);
  cvt_bf16_kernel<<<2048, 256, 0, stream>>>(x,  xb,  NTOK * HID / 4);
  cvt_bf16_kernel<<<2048, 256, 0, stream>>>(w1, w1b, NUM_EXPERTS * INTER * HID / 4);
  cvt_bf16_kernel<<<2048, 256, 0, stream>>>(w2, w2b, NUM_EXPERTS * HID * INTER / 4);
  router_kernel<<<NTOK / 8, 256, 0, stream>>>(x, gw, counts, tok_list, wgt_list,
                                              logits_out);
  const size_t smem = (size_t)BT * HROW * 2 + BT * 12 + 256;   // ~132 KB LDS
  moe_expert_kernel<<<NUM_EXPERTS * BLK_PER_E, 256, smem, stream>>>(
      xb, w1b, w2b, b1, b2, counts, tok_list, wgt_list, y_slot);
  finalize_kernel<<<2048, 256, 0, stream>>>(y_slot, out);
}
